// AttentionBlock_85658827752065
// MI455X (gfx1250) — compile-verified, matrix operand order checked by reading, not run on hardware
//
#include <hip/hip_runtime.h>

// ---------------------------------------------------------------------------
// AttentionBlock for MI455X (gfx1250): GroupNorm + QKV + 1-head flash attn +
// proj + residual.  All GEMMs in bf16 via v_wmma_f32_16x16x32_bf16 (wave32).
// Attention K/V tiles staged block-wide into LDS with the CDNA5 async path
// (global_load_async_to_lds_b128 / s_wait_asynccnt), fragments via ds_load.
// B=4, C=256, H=W=64 -> N=4096, groups=32, d=256, scale=1/16.
// ---------------------------------------------------------------------------

typedef __bf16 bf16;
typedef bf16  bf16x8 __attribute__((ext_vector_type(8)));
typedef bf16  v16bf  __attribute__((ext_vector_type(16)));
typedef float v8f    __attribute__((ext_vector_type(8)));

#define BATCH 4
#define CH    256
#define NSP   4096      // H*W
#define NGRP  32
#define CPG   8         // channels per group

#define KSTRIDE 264     // K tile LDS row stride (32 rows x 256 + 8 pad, bf16)
#define VSTRIDE 40      // V tile LDS row stride (256 rows x 32 + 8 pad, bf16)

__device__ __forceinline__ v8f zero8() {
    v8f z = {0.f,0.f,0.f,0.f,0.f,0.f,0.f,0.f};
    return z;
}

// Load one 16-bit A/B WMMA fragment (16 bf16 per lane) from a row whose K
// window starts at `base`.  Per ISA 16-bit layout: element i holds
// K = (i<8?0:16) + half*8 + (i%8)  -> two contiguous 16B chunks.
__device__ __forceinline__ v16bf ldfrag(const bf16* base, int half) {
    const bf16* p = base + half * 8;
    bf16x8 lo = *(const bf16x8*)(p);
    bf16x8 hi = *(const bf16x8*)(p + 16);
    v16bf r;
#pragma unroll
    for (int i = 0; i < 8; ++i) { r[i] = lo[i]; r[i + 8] = hi[i]; }
    return r;
}

__device__ __forceinline__ v8f wmma_bf16(v16bf a, v16bf b, v8f c) {
    return __builtin_amdgcn_wmma_f32_16x16x32_bf16(
        /*neg_a=*/false, a, /*neg_b=*/false, b,
        /*c_mod=*/(short)0, c, /*reuse_a=*/false, /*reuse_b=*/false);
}

// CDNA5 async copy: 16 bytes global -> LDS, tracked by ASYNCcnt.
__device__ __forceinline__ void async_b128(unsigned lds_off, const bf16* g) {
    asm volatile("global_load_async_to_lds_b128 %0, %1, off"
                 :: "v"(lds_off), "v"(g) : "memory");
}
__device__ __forceinline__ void wait_async0() {
    asm volatile("s_wait_asynccnt 0" ::: "memory");
}

// ---------------------------------------------------------------------------
// 1) GroupNorm statistics: one block per (b,g); 8*4096 elements reduced.
// ---------------------------------------------------------------------------
__global__ void gn_stats_kernel(const float* __restrict__ x,
                                float* __restrict__ stats) {
    __shared__ float shs[256], shq[256];
    const int bg  = blockIdx.x;            // b*32+g
    const int tid = threadIdx.x;
    const float* p = x + (size_t)bg * CPG * NSP;
    float s = 0.f, q = 0.f;
    for (int i = tid; i < CPG * NSP; i += 256) {
        float v = p[i];
        s += v; q += v * v;
    }
    shs[tid] = s; shq[tid] = q;
    __syncthreads();
    for (int st = 128; st > 0; st >>= 1) {
        if (tid < st) { shs[tid] += shs[tid + st]; shq[tid] += shq[tid + st]; }
        __syncthreads();
    }
    if (tid == 0) {
        const float inv_n = 1.f / (float)(CPG * NSP);
        float mean = shs[0] * inv_n;
        float var  = shq[0] * inv_n - mean * mean;
        stats[bg * 2 + 0] = mean;
        stats[bg * 2 + 1] = rsqrtf(var + 1e-5f);
    }
}

// ---------------------------------------------------------------------------
// 2) GroupNorm apply + transpose + cast: xnT[b][n][c] (bf16, B-matrix layout)
// ---------------------------------------------------------------------------
__global__ void gn_apply_kernel(const float* __restrict__ x,
                                const float* __restrict__ stats,
                                const float* __restrict__ w,
                                const float* __restrict__ bia,
                                bf16* __restrict__ xnt) {
    size_t i = (size_t)blockIdx.x * 256 + threadIdx.x;   // over B*C*N
    int n = (int)(i & (NSP - 1));
    size_t t = i >> 12;
    int c = (int)(t & (CH - 1));
    int b = (int)(t >> 8);
    int g = c >> 3;
    float mean = stats[(b * NGRP + g) * 2 + 0];
    float rstd = stats[(b * NGRP + g) * 2 + 1];
    float v = (x[i] - mean) * rstd * w[c] + bia[c];
    xnt[((size_t)b * NSP + n) * CH + c] = (bf16)v;
}

// ---------------------------------------------------------------------------
// 3) Cast weights to bf16 (row-major, A-matrix friendly)
// ---------------------------------------------------------------------------
__global__ void cast_w_kernel(const float* __restrict__ qw,
                              const float* __restrict__ pw,
                              bf16* __restrict__ qwb,
                              bf16* __restrict__ pwb) {
    int i = blockIdx.x * 256 + threadIdx.x;     // 0 .. 768*256-1
    if (i < 3 * CH * CH) qwb[i] = (bf16)qw[i];
    if (i < CH * CH)     pwb[i] = (bf16)pw[i];
}

// ---------------------------------------------------------------------------
// 4) QKV GEMM: qkv[o][n] = sum_c W[o][c]*xnT[n][c] + bias.
//    Epilogue scatters QT (scaled), KT ([n][d]) and V ([d][n]) in bf16.
//    One wave per 16(o) x 64(n) tile; 4 waves per block.
// ---------------------------------------------------------------------------
__global__ void qkv_gemm_kernel(const bf16* __restrict__ W,
                                const bf16* __restrict__ Xt,
                                const float* __restrict__ bias,
                                bf16* __restrict__ Qt,
                                bf16* __restrict__ Kt,
                                bf16* __restrict__ V) {
    const int lane = threadIdx.x & 31;
    const int wave = threadIdx.x >> 5;
    const int l16  = lane & 15;
    const int half = lane >> 4;
    int task = blockIdx.x * 4 + wave;            // B*48*64 tasks
    int b    = task / (48 * 64);
    int rem  = task % (48 * 64);
    int ot   = rem / 64;                         // o-tile (16 rows)
    int n0   = (rem % 64) * 64;                  // n strip of 64

    const bf16* Wrow = W + (size_t)(ot * 16 + l16) * CH;
    const bf16* Xb   = Xt + (size_t)b * NSP * CH;

    v8f acc[4];
#pragma unroll
    for (int nt = 0; nt < 4; ++nt) acc[nt] = zero8();

#pragma unroll
    for (int c0 = 0; c0 < CH; c0 += 32) {
        v16bf a = ldfrag(Wrow + c0, half);
#pragma unroll
        for (int nt = 0; nt < 4; ++nt) {
            v16bf bmat = ldfrag(Xb + (size_t)(n0 + nt * 16 + l16) * CH + c0, half);
            acc[nt] = wmma_bf16(a, bmat, acc[nt]);
        }
    }

    const int o0 = ot * 16;
    const float qscale = 0.0625f;               // d^-0.5, d=256
#pragma unroll
    for (int nt = 0; nt < 4; ++nt) {
#pragma unroll
        for (int r = 0; r < 8; ++r) {
            int m = o0 + r + 8 * half;          // output channel (0..767)
            int n = n0 + nt * 16 + l16;
            float val = acc[nt][r] + bias[m];
            if (o0 < CH) {
                Qt[((size_t)b * NSP + n) * CH + m] = (bf16)(val * qscale);
            } else if (o0 < 2 * CH) {
                Kt[((size_t)b * NSP + n) * CH + (m - CH)] = (bf16)val;
            } else {
                V[((size_t)b * CH + (m - 2 * CH)) * NSP + n] = (bf16)val;
            }
        }
    }
}

// ---------------------------------------------------------------------------
// 5) Flash attention: S = QT*K (q pre-scaled), online softmax, O += P*V.
//    Block = 4 waves; wave owns 16 query rows; j-tiles of 32.
//    K tile (32x256) and V tile (256x32) staged block-wide into LDS with
//    global_load_async_to_lds_b128; fragments read from LDS (ds_load_b128).
//    Q fragments are j-invariant and kept in registers.
//    Output AOt[b][i][d] (bf16) = B-matrix layout for the projection GEMM.
// ---------------------------------------------------------------------------
__global__ void __launch_bounds__(128, 1)
attn_kernel(const bf16* __restrict__ Qt,
            const bf16* __restrict__ Kt,
            const bf16* __restrict__ V,
            bf16* __restrict__ AOt) {
    __shared__ __align__(16) bf16 Klds[32 * KSTRIDE];    // 16.9 KB
    __shared__ __align__(16) bf16 Vlds[CH * VSTRIDE];    // 20.5 KB
    __shared__ __align__(16) bf16 pscr[4][16 * 32];      // 4 KB P transpose
    const int tid  = threadIdx.x;
    const int lane = tid & 31;
    const int wave = tid >> 5;
    const int l16  = lane & 15;
    const int half = lane >> 4;

    const int b  = blockIdx.x >> 6;             // 64 blocks per batch
    const int i0 = (blockIdx.x & 63) * 64 + wave * 16;

    const bf16* Qrow = Qt + ((size_t)b * NSP + i0 + l16) * CH;   // A base row
    const bf16* Kb   = Kt + (size_t)b * NSP * CH;
    const bf16* Vb   = V  + (size_t)b * CH * NSP;
    bf16* myp = pscr[wave];

    // Hoist the 8 Q A-fragments (j-invariant).
    v16bf qa[8];
#pragma unroll
    for (int c = 0; c < 8; ++c) qa[c] = ldfrag(Qrow + c * 32, half);

    v8f acc[16];
#pragma unroll
    for (int dt = 0; dt < 16; ++dt) acc[dt] = zero8();
    float rmax[8], rsum[8];
#pragma unroll
    for (int r = 0; r < 8; ++r) { rmax[r] = -3.0e38f; rsum[r] = 0.f; }

    for (int j0 = 0; j0 < NSP; j0 += 32) {
        __syncthreads();                        // prev-iter LDS reads done
        // ---- stage K (32 x 256) and V (256 x 32) tiles: 16B per async op --
#pragma unroll
        for (int k = 0; k < 8; ++k) {           // K: 1024 chunks / 128 thr
            int c   = k * 128 + tid;
            int row = c >> 5, col = c & 31;
            async_b128((unsigned)(size_t)&Klds[row * KSTRIDE + col * 8],
                       Kb + (size_t)(j0 + row) * CH + col * 8);
        }
#pragma unroll
        for (int k = 0; k < 8; ++k) {           // V: 1024 chunks / 128 thr
            int c = k * 128 + tid;
            int d = c >> 2, col = c & 3;
            async_b128((unsigned)(size_t)&Vlds[d * VSTRIDE + col * 8],
                       Vb + (size_t)d * NSP + j0 + col * 8);
        }
        wait_async0();
        __syncthreads();                        // all tiles visible

        // ---- S tile (16 x 32): contraction over d=256, B frags from LDS ---
        const bf16* Kr0 = &Klds[(size_t)l16 * KSTRIDE];
        const bf16* Kr1 = &Klds[(size_t)(16 + l16) * KSTRIDE];
        v8f s0 = zero8(), s1 = zero8();
#pragma unroll
        for (int c = 0; c < 8; ++c) {
            v16bf b0 = ldfrag(Kr0 + c * 32, half);
            v16bf b1 = ldfrag(Kr1 + c * 32, half);
            s0 = wmma_bf16(qa[c], b0, s0);
            s1 = wmma_bf16(qa[c], b1, s1);
        }
        // ---- online softmax (rows r and r+8 live in lane halves) ----
#pragma unroll
        for (int r = 0; r < 8; ++r) {
            float v0 = s0[r], v1 = s1[r];
            float tmax = fmaxf(v0, v1);
#pragma unroll
            for (int off = 1; off < 16; off <<= 1)
                tmax = fmaxf(tmax, __shfl_xor(tmax, off, 16));
            float newm = fmaxf(rmax[r], tmax);
            float corr = __expf(rmax[r] - newm);
            rmax[r] = newm;
            float p0 = __expf(v0 - newm);
            float p1 = __expf(v1 - newm);
            float ps = p0 + p1;
#pragma unroll
            for (int off = 1; off < 16; off <<= 1)
                ps += __shfl_xor(ps, off, 16);
            rsum[r] = rsum[r] * corr + ps;
#pragma unroll
            for (int dt = 0; dt < 16; ++dt) acc[dt][r] *= corr;
            int mrow = r + 8 * half;            // D-layout row
            myp[mrow * 32 + l16]      = (bf16)p0;
            myp[mrow * 32 + 16 + l16] = (bf16)p1;
        }
        // same-wave LDS ops are in-order; keep the compiler honest
        asm volatile("" ::: "memory");
        // ---- O += P * V : A frag from LDS scratch, B frags from V tile ----
        v16bf pa = ldfrag(myp + l16 * 32, half);
#pragma unroll
        for (int dt = 0; dt < 16; ++dt) {
            v16bf vb = ldfrag(&Vlds[(size_t)(dt * 16 + l16) * VSTRIDE], half);
            acc[dt] = wmma_bf16(pa, vb, acc[dt]);
        }
    }

    // ---- normalize and store AOt[b][i][d] ----
#pragma unroll
    for (int r = 0; r < 8; ++r) {
        float inv = 1.f / rsum[r];
        int row = i0 + r + 8 * half;
#pragma unroll
        for (int dt = 0; dt < 16; ++dt)
            AOt[((size_t)b * NSP + row) * CH + dt * 16 + l16] =
                (bf16)(acc[dt][r] * inv);
    }
}

// ---------------------------------------------------------------------------
// 6) Projection GEMM + bias + residual: out = x + proj_w * attn_out + proj_b
// ---------------------------------------------------------------------------
__global__ void proj_gemm_kernel(const bf16* __restrict__ W,
                                 const bf16* __restrict__ AOt,
                                 const float* __restrict__ bias,
                                 const float* __restrict__ x,
                                 float* __restrict__ out) {
    const int lane = threadIdx.x & 31;
    const int wave = threadIdx.x >> 5;
    const int l16  = lane & 15;
    const int half = lane >> 4;
    int task = blockIdx.x * 4 + wave;            // B*16*64 tasks
    int b    = task / (16 * 64);
    int rem  = task % (16 * 64);
    int ot   = rem / 64;
    int n0   = (rem % 64) * 64;

    const bf16* Wrow = W + (size_t)(ot * 16 + l16) * CH;
    const bf16* Ab   = AOt + (size_t)b * NSP * CH;

    v8f acc[4];
#pragma unroll
    for (int nt = 0; nt < 4; ++nt) acc[nt] = zero8();

#pragma unroll
    for (int c0 = 0; c0 < CH; c0 += 32) {
        v16bf a = ldfrag(Wrow + c0, half);
#pragma unroll
        for (int nt = 0; nt < 4; ++nt) {
            v16bf bmat = ldfrag(Ab + (size_t)(n0 + nt * 16 + l16) * CH + c0, half);
            acc[nt] = wmma_bf16(a, bmat, acc[nt]);
        }
    }

    const int o0 = ot * 16;
#pragma unroll
    for (int nt = 0; nt < 4; ++nt) {
#pragma unroll
        for (int r = 0; r < 8; ++r) {
            int m = o0 + r + 8 * half;
            int n = n0 + nt * 16 + l16;
            size_t idx = ((size_t)b * CH + m) * NSP + n;
            out[idx] = x[idx] + bias[m] + acc[nt][r];
        }
    }
}

// ---------------------------------------------------------------------------
// Launch
// ---------------------------------------------------------------------------
extern "C" void kernel_launch(void* const* d_in, const int* in_sizes, int n_in,
                              void* d_out, int out_size, void* d_ws, size_t ws_size,
                              hipStream_t stream) {
    const float* x      = (const float*)d_in[0];
    const float* norm_w = (const float*)d_in[1];
    const float* norm_b = (const float*)d_in[2];
    const float* qkv_w  = (const float*)d_in[3];
    const float* qkv_b  = (const float*)d_in[4];
    const float* proj_w = (const float*)d_in[5];
    const float* proj_b = (const float*)d_in[6];
    float* out = (float*)d_out;

    char* ws = (char*)d_ws;
    size_t off = 0;
    float* stats = (float*)(ws + off); off += 1024;                       // B*32*2 f32
    bf16*  qwb   = (bf16*)(ws + off);  off += (size_t)3*CH*CH*2;          // 384 KB
    bf16*  pwb   = (bf16*)(ws + off);  off += (size_t)CH*CH*2;            // 128 KB
    bf16*  Xt    = (bf16*)(ws + off);  off += (size_t)BATCH*NSP*CH*2;     // 8 MB
    bf16*  Qt    = (bf16*)(ws + off);  off += (size_t)BATCH*NSP*CH*2;
    bf16*  Kt    = (bf16*)(ws + off);  off += (size_t)BATCH*NSP*CH*2;
    bf16*  V     = (bf16*)(ws + off);  off += (size_t)BATCH*CH*NSP*2;
    bf16*  AOt   = (bf16*)(ws + off);  off += (size_t)BATCH*NSP*CH*2;

    gn_stats_kernel<<<BATCH * NGRP, 256, 0, stream>>>(x, stats);
    gn_apply_kernel<<<(BATCH * CH * NSP) / 256, 256, 0, stream>>>(
        x, stats, norm_w, norm_b, Xt);
    cast_w_kernel<<<(3 * CH * CH) / 256, 256, 0, stream>>>(qkv_w, proj_w, qwb, pwb);
    qkv_gemm_kernel<<<(BATCH * 48 * 64) / 4, 128, 0, stream>>>(
        qwb, Xt, qkv_b, Qt, Kt, V);
    attn_kernel<<<BATCH * (NSP / 64), 128, 0, stream>>>(Qt, Kt, V, AOt);
    proj_gemm_kernel<<<(BATCH * 16 * 64) / 4, 128, 0, stream>>>(
        pwb, AOt, proj_b, x, out);
}
